// MemoryUpdater_44152263803424
// MI455X (gfx1250) — matrix operand order repaired and checked by reading, not compiled, not run on hardware
//
#include <hip/hip_runtime.h>

// ---------------------------------------------------------------------------
// Types / helpers
// ---------------------------------------------------------------------------
typedef __bf16 bf16x16 __attribute__((ext_vector_type(16)));
typedef float  f32x8   __attribute__((ext_vector_type(8)));

__device__ __forceinline__ unsigned short f2bf(float x) {
  unsigned u = __float_as_uint(x);
  u += 0x7FFFu + ((u >> 16) & 1u);          // round-to-nearest-even
  return (unsigned short)(u >> 16);
}
__device__ __forceinline__ float bf2f(unsigned short b) {
  return __uint_as_float(((unsigned)b) << 16);
}
__device__ __forceinline__ unsigned pack2(unsigned short a, unsigned short b) {
  return (unsigned)a | ((unsigned)b << 16);
}
__device__ __forceinline__ bf16x16 frag(uint4 a, uint4 b) {
  union { uint4 u[2]; bf16x16 v; } f; f.u[0] = a; f.u[1] = b; return f.v;
}
__device__ __forceinline__ f32x8 mma(bf16x16 a, bf16x16 b, f32x8 c) {
  return __builtin_amdgcn_wmma_f32_16x16x32_bf16(false, a, false, b,
                                                 (short)0, c, false, false);
}
__device__ __forceinline__ float sigmoidf_(float x) { return 1.f / (1.f + __expf(-x)); }
__device__ __forceinline__ float tanhf_(float x)    { return 2.f / (1.f + __expf(-2.f * x)) - 1.f; }

// ---------------------------------------------------------------------------
// Problem constants
// ---------------------------------------------------------------------------
#define N_NODES 1000000
#define MEM_DIM 128
#define MSG_DIM 256
#define BATCH   200000

// LDS strides (padded, in elements)
#define MSTR 264   // 256 + 8 bf16  (528 B row)
#define HSTR 136   // 128 + 8 bf16  (272 B row)

// msg hi/lo + h hi/lo + nids  = 67584 + 34816 + 256 = 102656 B  (3 WGs/WGP)
#define SMEM_BYTES (2*64*MSTR*2 + 2*64*HSTR*2 + 64*4)

// ---------------------------------------------------------------------------
// Weight precision-split: f32 -> bf16 hi + bf16 lo
// ---------------------------------------------------------------------------
__global__ void prep_weights(const float* __restrict__ wih, const float* __restrict__ whh,
                             unsigned short* __restrict__ wih_h, unsigned short* __restrict__ wih_l,
                             unsigned short* __restrict__ whh_h, unsigned short* __restrict__ whh_l) {
  int i = blockIdx.x * 256 + threadIdx.x;
  if (i < 3 * MEM_DIM * MSG_DIM) {           // 98304
    float x = wih[i];
    unsigned short h = f2bf(x);
    wih_h[i] = h;
    wih_l[i] = f2bf(x - bf2f(h));
  }
  if (i < 3 * MEM_DIM * MEM_DIM) {           // 49152
    float x = whh[i];
    unsigned short h = f2bf(x);
    whh_h[i] = h;
    whh_l[i] = f2bf(x - bf2f(h));
  }
}

// ---------------------------------------------------------------------------
// Bulk clone (float4 grid-stride)
// ---------------------------------------------------------------------------
__global__ void copy_f4(const float4* __restrict__ src, float4* __restrict__ dst, int n) {
  int stride = gridDim.x * blockDim.x;
  for (int i = blockIdx.x * blockDim.x + threadIdx.x; i < n; i += stride)
    dst[i] = src[i];
}

__global__ void scatter_lu(const int* __restrict__ ids, const float* __restrict__ ts,
                           float* __restrict__ lu, int n) {
  int i = blockIdx.x * blockDim.x + threadIdx.x;
  if (i < n) lu[ids[i]] = ts[i];
}

// ---------------------------------------------------------------------------
// GRU memory update: 4 waves/block, 16 batch rows per wave, bf16x3 WMMA GEMMs
// ---------------------------------------------------------------------------
__global__ void gru_kernel(const float* __restrict__ messages,
                           const float* __restrict__ memory,
                           const int*   __restrict__ node_ids,
                           const float* __restrict__ b_ih,
                           const float* __restrict__ b_hh,
                           const unsigned short* __restrict__ wih_h,
                           const unsigned short* __restrict__ wih_l,
                           const unsigned short* __restrict__ whh_h,
                           const unsigned short* __restrict__ whh_l,
                           float* __restrict__ out_mem) {
  extern __shared__ char smem[];
  unsigned short* msg_hi = (unsigned short*)smem;          // 64 x MSTR
  unsigned short* msg_lo = msg_hi + 64 * MSTR;
  unsigned short* h_hi   = msg_lo + 64 * MSTR;             // 64 x HSTR
  unsigned short* h_lo   = h_hi + 64 * HSTR;
  int*            nids   = (int*)(h_lo + 64 * HSTR);       // 64

  const int tid = threadIdx.x;
  const int M0  = blockIdx.x * 64;                         // block-base batch row

  // ---- stage node ids ----
  if (tid < 64) nids[tid] = node_ids[M0 + tid];

  // ---- stage messages as bf16 hi/lo (64 rows x 256) ----
  const float4* msg4 = (const float4*)messages;
  for (int i = tid; i < 64 * 64; i += 128) {
    int row = i >> 6, c4 = i & 63, c = c4 << 2;
    float4 v = msg4[(size_t)(M0 + row) * 64 + c4];
    unsigned short h0 = f2bf(v.x), h1 = f2bf(v.y), h2 = f2bf(v.z), h3 = f2bf(v.w);
    unsigned short l0 = f2bf(v.x - bf2f(h0)), l1 = f2bf(v.y - bf2f(h1));
    unsigned short l2 = f2bf(v.z - bf2f(h2)), l3 = f2bf(v.w - bf2f(h3));
    *(uint2*)&msg_hi[row * MSTR + c] = make_uint2(pack2(h0, h1), pack2(h2, h3));
    *(uint2*)&msg_lo[row * MSTR + c] = make_uint2(pack2(l0, l1), pack2(l2, l3));
  }

  // ---- gather h rows as bf16 hi/lo (hi+lo reconstructs ~f32, err ~7e-6) ----
  const float4* mem4 = (const float4*)memory;
  for (int i = tid; i < 64 * 32; i += 128) {
    int row = i >> 5, c4 = i & 31, c = c4 << 2;
    int nid = node_ids[M0 + row];
    float4 v = mem4[(size_t)nid * 32 + c4];
    unsigned short h0 = f2bf(v.x), h1 = f2bf(v.y), h2 = f2bf(v.z), h3 = f2bf(v.w);
    unsigned short l0 = f2bf(v.x - bf2f(h0)), l1 = f2bf(v.y - bf2f(h1));
    unsigned short l2 = f2bf(v.z - bf2f(h2)), l3 = f2bf(v.w - bf2f(h3));
    *(uint2*)&h_hi[row * HSTR + c] = make_uint2(pack2(h0, h1), pack2(h2, h3));
    *(uint2*)&h_lo[row * HSTR + c] = make_uint2(pack2(l0, l1), pack2(l2, l3));
  }
  __syncthreads();

  // ---- per-wave 16x16 WMMA tiles ----
  const int wv   = tid >> 5;
  const int lane = tid & 31;
  const int ln   = lane & 15;       // column within tile (B/C) or A row
  const int h16  = lane >> 4;       // half-wave select

  const unsigned short* mhA = msg_hi + (wv * 16 + ln) * MSTR;  // A-frag row base
  const unsigned short* mlA = msg_lo + (wv * 16 + ln) * MSTR;
  const unsigned short* hhA = h_hi   + (wv * 16 + ln) * HSTR;
  const unsigned short* hlA = h_lo   + (wv * 16 + ln) * HSTR;

#pragma unroll 1
  for (int dt = 0; dt < 8; ++dt) {                 // 8 d-tiles of 16 columns
    // Fence: keep A-fragments in LDS (re-load per dt) instead of letting LLVM
    // hoist ~192 VGPRs of loop-invariant fragments and spill them to scratch.
    asm volatile("" ::: "memory");

    f32x8 acc[6];                                  // 0..2 = gi r,z,n ; 3..5 = gh r,z,n
#pragma unroll
    for (int t = 0; t < 6; ++t) acc[t] = (f32x8){0.f,0.f,0.f,0.f,0.f,0.f,0.f,0.f};

    // --- gi = messages @ W_ih.T  (K = 256) ---
#pragma unroll
    for (int kf = 0; kf < 8; ++kf) {
      int ka = kf * 32 + h16 * 8;                  // A: 8+8 K-chunks per lane
      bf16x16 Ah = frag(*(const uint4*)(mhA + ka), *(const uint4*)(mhA + ka + 16));
      bf16x16 Al = frag(*(const uint4*)(mlA + ka), *(const uint4*)(mlA + ka + 16));
      int kb = kf * 32 + h16 * 8;                  // B: 8+8 K-chunks per lane
#pragma unroll
      for (int g = 0; g < 3; ++g) {
        size_t woff = (size_t)(g * 128 + dt * 16 + ln) * 256 + kb;
        const unsigned short* wr = wih_h + woff;
        const unsigned short* wl = wih_l + woff;
        bf16x16 Bh = frag(*(const uint4*)wr, *(const uint4*)(wr + 16));
        bf16x16 Bl = frag(*(const uint4*)wl, *(const uint4*)(wl + 16));
        acc[g] = mma(Ah, Bh, acc[g]);
        acc[g] = mma(Al, Bh, acc[g]);
        acc[g] = mma(Ah, Bl, acc[g]);
      }
    }

    // --- gh = h @ W_hh.T  (K = 128) ---
#pragma unroll
    for (int kf = 0; kf < 4; ++kf) {
      int ka = kf * 32 + h16 * 8;
      bf16x16 Ah = frag(*(const uint4*)(hhA + ka), *(const uint4*)(hhA + ka + 16));
      bf16x16 Al = frag(*(const uint4*)(hlA + ka), *(const uint4*)(hlA + ka + 16));
      int kb = kf * 32 + h16 * 8;
#pragma unroll
      for (int g = 0; g < 3; ++g) {
        size_t woff = (size_t)(g * 128 + dt * 16 + ln) * 128 + kb;
        const unsigned short* wr = whh_h + woff;
        const unsigned short* wl = whh_l + woff;
        bf16x16 Bh = frag(*(const uint4*)wr, *(const uint4*)(wr + 16));
        bf16x16 Bl = frag(*(const uint4*)wl, *(const uint4*)(wl + 16));
        acc[3 + g] = mma(Ah, Bh, acc[3 + g]);
        acc[3 + g] = mma(Al, Bh, acc[3 + g]);
        acc[3 + g] = mma(Ah, Bl, acc[3 + g]);
      }
    }

    // --- gates + scatter (C layout: row = j + h16*8, col = ln) ---
    int dcol = dt * 16 + ln;
    float bir = b_ih[dcol], biz = b_ih[128 + dcol], bin = b_ih[256 + dcol];
    float bhr = b_hh[dcol], bhz = b_hh[128 + dcol], bhn = b_hh[256 + dcol];
#pragma unroll
    for (int j = 0; j < 8; ++j) {
      int ml = wv * 16 + j + h16 * 8;              // block-local batch row
      float hv = bf2f(h_hi[ml * HSTR + dcol]) + bf2f(h_lo[ml * HSTR + dcol]);
      float r  = sigmoidf_(acc[0][j] + bir + acc[3][j] + bhr);
      float z  = sigmoidf_(acc[1][j] + biz + acc[4][j] + bhz);
      float nn = tanhf_(acc[2][j] + bin + r * (acc[5][j] + bhn));
      float o  = nn + z * (hv - nn);
      out_mem[(size_t)nids[ml] * MEM_DIM + dcol] = o;
    }
  }
}

// ---------------------------------------------------------------------------
// Launch
// ---------------------------------------------------------------------------
extern "C" void kernel_launch(void* const* d_in, const int* in_sizes, int n_in,
                              void* d_out, int out_size, void* d_ws, size_t ws_size,
                              hipStream_t stream) {
  const float* memory      = (const float*)d_in[0];
  const float* last_update = (const float*)d_in[1];
  const float* messages    = (const float*)d_in[2];
  const float* timestamps  = (const float*)d_in[3];
  const float* W_ih        = (const float*)d_in[4];
  const float* W_hh        = (const float*)d_in[5];
  const float* b_ih        = (const float*)d_in[6];
  const float* b_hh        = (const float*)d_in[7];
  const int*   node_ids    = (const int*)d_in[8];

  float* out_mem = (float*)d_out;
  float* out_lu  = out_mem + (size_t)N_NODES * MEM_DIM;

  unsigned short* wih_h = (unsigned short*)d_ws;
  unsigned short* wih_l = wih_h + 3 * MEM_DIM * MSG_DIM;   // 98304
  unsigned short* whh_h = wih_l + 3 * MEM_DIM * MSG_DIM;
  unsigned short* whh_l = whh_h + 3 * MEM_DIM * MEM_DIM;   // 49152

  // 1) split weights to bf16 hi/lo (tiny, L2-resident afterwards)
  prep_weights<<<(3 * MEM_DIM * MSG_DIM + 255) / 256, 256, 0, stream>>>(
      W_ih, W_hh, wih_h, wih_l, whh_h, whh_l);

  // 2) clone memory and last_update into the output
  copy_f4<<<8192, 256, 0, stream>>>((const float4*)memory, (float4*)out_mem,
                                    N_NODES * MEM_DIM / 4);
  copy_f4<<<1024, 256, 0, stream>>>((const float4*)last_update, (float4*)out_lu,
                                    N_NODES / 4);
  scatter_lu<<<(BATCH + 255) / 256, 256, 0, stream>>>(node_ids, timestamps, out_lu, BATCH);

  // 3) GRU cell + scatter of new_h (overwrites cloned rows; same-stream ordering)
  gru_kernel<<<BATCH / 64, 128, SMEM_BYTES, stream>>>(
      messages, memory, node_ids, b_ih, b_hh,
      wih_h, wih_l, whh_h, whh_l, out_mem);
}
